// DDGPT_17248588661280
// MI455X (gfx1250) — matrix operand order repaired and checked by reading, not compiled, not run on hardware
//
#include <hip/hip_runtime.h>
#include <math.h>

// Problem dims (fixed by reference)
#define BB   2
#define TT   1024
#define VV   32000
#define EE   768
#define HH   12
#define HD   64
#define LLAYERS 4
#define ROWS (BB*TT)          // 2048
#define E3   (3*EE)           // 2304

typedef __attribute__((ext_vector_type(16))) __bf16 bf16x16;
typedef __attribute__((ext_vector_type(8)))  float  f32x8;
typedef int v4i_vs __attribute__((vector_size(16)));   // async-LDS builtin param type

union BFrag { bf16x16 v; unsigned u[8]; };

// 16-bit A-matrix 16x32 fill (contiguous k, matching B): lanes<16 hold K=0..15,
// lanes 16..31 hold K=16..31; pair p -> starting (even) k.
__device__ __forceinline__ int a_kmap(int p, int half) {
  int k = 2 * p;
  return k + (half ? 16 : 0);
}

// ---------------------------------------------------------------------------
// 16-byte stage: async direct-to-LDS if the toolchain exposes it, else
// through-VGPR copy. Guarded so either way compiles.
// ---------------------------------------------------------------------------
#if __has_builtin(__builtin_amdgcn_global_load_async_to_lds_b128)
#define HAS_ASYNC_LDS 1
#else
#define HAS_ASYNC_LDS 0
#endif

__device__ __forceinline__ void stage16(const __bf16* g, __bf16* l) {
#if HAS_ASYNC_LDS
  __builtin_amdgcn_global_load_async_to_lds_b128(
      (v4i_vs*)(void*)g, (v4i_vs*)(void*)l, 0, 0);
#else
  *reinterpret_cast<uint4*>(l) = *reinterpret_cast<const uint4*>(g);
#endif
}

__device__ __forceinline__ void wait_stage() {
#if HAS_ASYNC_LDS
#if __has_builtin(__builtin_amdgcn_s_wait_asynccnt)
  __builtin_amdgcn_s_wait_asynccnt(0);
#else
  asm volatile("s_wait_asynccnt 0" ::: "memory");
#endif
#endif
}

// ---------------------------------------------------------------------------
// bf16 WMMA GEMM: C[M,N] = A[M,K] @ Bt^T (+bias) (gelu?) (+residual)
// A row-major bf16 [M][K]; Bt is B pre-transposed: [N][K] bf16.
// Workgroup tile 128(M) x 256(N) x 32(K); 8 waves (2x4), wave tile 64x64
// = 4x4 wmma blocks = 16 v_wmma per k-step. Double-buffered LDS with
// async global->LDS staging overlapping the WMMA work of the prior tile.
// ---------------------------------------------------------------------------
#define GTM 128
#define GTN 256
#define GTK 32
#define LSTR 40          // padded LDS row stride (halves): 80B rows
#define ASZ (GTM*LSTR)   // one A buffer (elems)
#define BSZ (GTN*LSTR)   // one B buffer (elems)

__global__ __launch_bounds__(256) void gemm_bf16_wmma(
    const __bf16* __restrict__ A, int lda,
    const __bf16* __restrict__ Bt, int ldb,   // ldb == K
    const float* __restrict__ bias,
    const float* __restrict__ residual, int ldr,
    float* __restrict__ outF, __bf16* __restrict__ outB,
    int M, int N, int K, int act)
{
  __shared__ __bf16 As[2 * ASZ];
  __shared__ __bf16 Bs[2 * BSZ];

  const int tid  = threadIdx.x;
  const int wid  = tid >> 5;
  const int lane = tid & 31;
  const int half = lane >> 4;
  const int r    = lane & 15;
  const int mBase = blockIdx.y * GTM;
  const int nBase = blockIdx.x * GTN;
  const int mWave = (wid >> 2) * 64;   // 2 waves along M
  const int nWave = (wid & 3) * 64;    // 4 waves along N

  // per-thread staging coordinates
  const int sArow = tid >> 1, sAch = (tid & 1) * 16;  // A: 128 rows x 32
  const int sBrow = tid;                               // B: 256 rows x 32

  f32x8 acc[4][4];
  for (int a0 = 0; a0 < 4; a0++)
    for (int b0 = 0; b0 < 4; b0++)
      for (int j = 0; j < 8; j++) acc[a0][b0][j] = 0.f;

  // prologue: stage k-tile 0 into buffer 0
  {
    const __bf16* ga = &A[(size_t)(mBase + sArow) * lda + sAch];
    stage16(ga,     &As[sArow * LSTR + sAch]);
    stage16(ga + 8, &As[sArow * LSTR + sAch + 8]);
    const __bf16* gb = &Bt[(size_t)(nBase + sBrow) * ldb];
    for (int j = 0; j < 4; j++)
      stage16(gb + 8 * j, &Bs[sBrow * LSTR + 8 * j]);
  }

  int buf = 0;
  for (int kt = 0; kt < K; kt += GTK) {
    wait_stage();        // own-wave async loads for 'buf' complete
    __syncthreads();     // all waves' loads complete + prior compute done

    if (kt + GTK < K) {  // stage next k-tile into the other buffer
      int nb2 = (buf ^ 1);
      const __bf16* ga = &A[(size_t)(mBase + sArow) * lda + kt + GTK + sAch];
      stage16(ga,     &As[nb2 * ASZ + sArow * LSTR + sAch]);
      stage16(ga + 8, &As[nb2 * ASZ + sArow * LSTR + sAch + 8]);
      const __bf16* gb = &Bt[(size_t)(nBase + sBrow) * ldb + kt + GTK];
      for (int j = 0; j < 4; j++)
        stage16(gb + 8 * j, &Bs[nb2 * BSZ + sBrow * LSTR + 8 * j]);
    }

    // fragments + 16 WMMAs on current buffer
    const __bf16* ab = &As[buf * ASZ];
    const __bf16* bb = &Bs[buf * BSZ];
    BFrag aF[4], bF[4];
    for (int mb = 0; mb < 4; mb++) {
      int row = mWave + mb * 16 + r;
      for (int p = 0; p < 8; p++)
        aF[mb].u[p] = *reinterpret_cast<const unsigned*>(&ab[row * LSTR + a_kmap(p, half)]);
    }
    for (int nb = 0; nb < 4; nb++) {
      int n = nWave + nb * 16 + r;
      for (int p = 0; p < 8; p++)
        bF[nb].u[p] = *reinterpret_cast<const unsigned*>(&bb[n * LSTR + half * 16 + 2 * p]);
    }
    for (int mb = 0; mb < 4; mb++)
      for (int nb = 0; nb < 4; nb++)
        acc[mb][nb] = __builtin_amdgcn_wmma_f32_16x16x32_bf16(
            false, aF[mb].v, false, bF[nb].v, (short)0, acc[mb][nb], false, false);
    buf ^= 1;
  }

  // epilogue: bias / gelu / residual / stores
  for (int mb = 0; mb < 4; mb++) {
    for (int nb = 0; nb < 4; nb++) {
      int col = nBase + nWave + nb * 16 + r;
      float bv = bias ? bias[col] : 0.f;
      for (int j = 0; j < 8; j++) {
        int row = mBase + mWave + mb * 16 + j + 8 * half;
        float v = acc[mb][nb][j] + bv;
        if (act == 1) {  // tanh gelu
          float c = v * v * v;
          v = 0.5f * v * (1.f + tanhf(0.7978845608f * (v + 0.044715f * c)));
        }
        if (residual) v += residual[(size_t)row * ldr + col];
        if (outF) outF[(size_t)row * (size_t)N + col] = v;
        if (outB) outB[(size_t)row * (size_t)N + col] = (__bf16)v;
      }
    }
  }
}

// ---------------------------------------------------------------------------
// Flash-style attention (no mask; scale = HD**0.5 = 8.0 per reference).
// Grid: (T/128, H, B); 256 threads = 8 waves, wave owns 16 q rows.
// ---------------------------------------------------------------------------
#define ASTR 72

__global__ __launch_bounds__(256) void attn_wmma(
    const __bf16* __restrict__ qkv, __bf16* __restrict__ y)
{
  __shared__ __bf16 Qs[128 * ASTR];
  __shared__ __bf16 Ks[64 * ASTR];
  __shared__ __bf16 Vt[64 * ASTR];   // transposed: Vt[d][key]
  __shared__ __bf16 Ps[128 * ASTR];  // per-wave 16-row slices

  const int tid  = threadIdx.x;
  const int wid  = tid >> 5;
  const int lane = tid & 31;
  const int half = lane >> 4;
  const int r    = lane & 15;
  const int qBlk = blockIdx.x * 128;
  const int h    = blockIdx.y;
  const int b    = blockIdx.z;
  const size_t rowBase = (size_t)b * TT;
  const int qRow0 = wid * 16;

  {  // stage Q (each wave loads exactly its own 16 rows)
    int row = tid >> 1, ch = (tid & 1) * 32;
    const uint4* g = reinterpret_cast<const uint4*>(
        &qkv[(rowBase + qBlk + row) * (size_t)E3 + h * HD + ch]);
    uint4* s = reinterpret_cast<uint4*>(&Qs[row * ASTR + ch]);
    s[0] = g[0]; s[1] = g[1]; s[2] = g[2]; s[3] = g[3];
  }

  float mrow[8], lrow[8];
  f32x8 oAcc[4];
  for (int j = 0; j < 8; j++) { mrow[j] = -1e30f; lrow[j] = 0.f; }
  for (int nb = 0; nb < 4; nb++)
    for (int j = 0; j < 8; j++) oAcc[nb][j] = 0.f;

  for (int kt = 0; kt < TT; kt += 64) {
    __syncthreads();
    {  // stage K tile (64 keys x 64 dim)
      int row = tid >> 2, dq = (tid & 3) * 16;
      const uint4* g = reinterpret_cast<const uint4*>(
          &qkv[(rowBase + kt + row) * (size_t)E3 + EE + h * HD + dq]);
      uint4* s = reinterpret_cast<uint4*>(&Ks[row * ASTR + dq]);
      s[0] = g[0]; s[1] = g[1];
    }
    {  // stage V tile transposed -> Vt[d][key]
      int key = tid >> 2, dq = (tid & 3) * 16;
      union { uint4 q[2]; __bf16 hh[16]; } tmp;
      const uint4* g = reinterpret_cast<const uint4*>(
          &qkv[(rowBase + kt + key) * (size_t)E3 + 2 * EE + h * HD + dq]);
      tmp.q[0] = g[0]; tmp.q[1] = g[1];
      for (int j = 0; j < 16; j++) Vt[(dq + j) * ASTR + key] = tmp.hh[j];
    }
    __syncthreads();

    // S = Q @ K^T (16x64 per wave; head dim 64 = 2 wmma k-steps)
    f32x8 sAcc[4];
    for (int nb = 0; nb < 4; nb++)
      for (int j = 0; j < 8; j++) sAcc[nb][j] = 0.f;
    for (int ks = 0; ks < 2; ks++) {
      BFrag aQ;
      for (int p = 0; p < 8; p++)
        aQ.u[p] = *reinterpret_cast<const unsigned*>(
            &Qs[(qRow0 + r) * ASTR + ks * 32 + a_kmap(p, half)]);
      for (int nb = 0; nb < 4; nb++) {
        BFrag bK;
        for (int p = 0; p < 8; p++)
          bK.u[p] = *reinterpret_cast<const unsigned*>(
              &Ks[(nb * 16 + r) * ASTR + ks * 32 + half * 16 + 2 * p]);
        sAcc[nb] = __builtin_amdgcn_wmma_f32_16x16x32_bf16(
            false, aQ.v, false, bK.v, (short)0, sAcc[nb], false, false);
      }
    }

    // scale + online softmax (16-lane row reductions; wave32)
    float mnew[8], scl[8], psum[8];
    for (int j = 0; j < 8; j++) {
      float mx = -1e30f;
      for (int nb = 0; nb < 4; nb++) {
        float v = sAcc[nb][j] * 8.0f;   // HD**0.5
        sAcc[nb][j] = v;
        mx = fmaxf(mx, v);
      }
      for (int off = 1; off < 16; off <<= 1)
        mx = fmaxf(mx, __shfl_xor(mx, off, 32));
      mnew[j] = fmaxf(mrow[j], mx);
      scl[j]  = __expf(mrow[j] - mnew[j]);
      psum[j] = 0.f;
    }
    for (int nb = 0; nb < 4; nb++)
      for (int j = 0; j < 8; j++) {
        float p = __expf(sAcc[nb][j] - mnew[j]);
        sAcc[nb][j] = p;
        psum[j] += p;
      }
    for (int j = 0; j < 8; j++) {
      float s = psum[j];
      for (int off = 1; off < 16; off <<= 1) s += __shfl_xor(s, off, 32);
      lrow[j] = lrow[j] * scl[j] + s;
      mrow[j] = mnew[j];
    }
    for (int nb = 0; nb < 4; nb++)
      for (int j = 0; j < 8; j++) oAcc[nb][j] *= scl[j];

    // restage P (C-layout -> A-fragment layout via per-wave LDS slice)
    for (int nb = 0; nb < 4; nb++)
      for (int j = 0; j < 8; j++)
        Ps[(qRow0 + j + 8 * half) * ASTR + nb * 16 + r] = (__bf16)sAcc[nb][j];

    // O += P @ V (64 keys = 2 wmma k-steps)
    for (int ks = 0; ks < 2; ks++) {
      BFrag aP;
      for (int p = 0; p < 8; p++)
        aP.u[p] = *reinterpret_cast<const unsigned*>(
            &Ps[(qRow0 + r) * ASTR + ks * 32 + a_kmap(p, half)]);
      for (int nb = 0; nb < 4; nb++) {
        BFrag bV;
        for (int p = 0; p < 8; p++)
          bV.u[p] = *reinterpret_cast<const unsigned*>(
              &Vt[(nb * 16 + r) * ASTR + ks * 32 + half * 16 + 2 * p]);
        oAcc[nb] = __builtin_amdgcn_wmma_f32_16x16x32_bf16(
            false, aP.v, false, bV.v, (short)0, oAcc[nb], false, false);
      }
    }
  }

  for (int nb = 0; nb < 4; nb++)
    for (int j = 0; j < 8; j++) {
      int row = qBlk + qRow0 + j + 8 * half;
      int col = h * HD + nb * 16 + r;
      y[(rowBase + row) * (size_t)EE + col] = (__bf16)(oAcc[nb][j] / lrow[j]);
    }
}

// ---------------------------------------------------------------------------
// Helper kernels
// ---------------------------------------------------------------------------
__global__ void cvt_f32_bf16(const float* __restrict__ src, __bf16* __restrict__ dst, long long n) {
  long long i = (long long)blockIdx.x * 256 + threadIdx.x;
  if (i < n) dst[i] = (__bf16)src[i];
}

// convert + transpose one weight per grid.z layer: src [K][N] f32 -> dst [N][K] bf16
__global__ __launch_bounds__(256) void cvtT_bf16(const float* __restrict__ src,
                                                 __bf16* __restrict__ dst, int K, int N) {
  const float* s = src + (size_t)blockIdx.z * K * N;
  __bf16* d = dst + (size_t)blockIdx.z * K * N;
  int n = blockIdx.x * 32 + (threadIdx.x & 31);
  int kBase = blockIdx.y * 32;
  for (int kk = threadIdx.x >> 5; kk < 32; kk += 8) {
    int k = kBase + kk;
    d[(size_t)n * K + k] = (__bf16)s[(size_t)k * N + n];
  }
}

__global__ void embed_k(const int* __restrict__ tok, const float* __restrict__ wte,
                        const float* __restrict__ wpe, float* __restrict__ x) {
  int row = blockIdx.x;
  int pos = row % TT;
  int t   = tok[row];
  for (int e = threadIdx.x; e < EE; e += 256)
    x[(size_t)row * EE + e] = wte[(size_t)t * EE + e] + wpe[(size_t)pos * EE + e];
}

__global__ __launch_bounds__(256) void ln_bf16(const float* __restrict__ x,
                                               const float* __restrict__ w,
                                               const float* __restrict__ b,
                                               __bf16* __restrict__ out) {
  __shared__ float red[256];
  int row = blockIdx.x, tid = threadIdx.x;
  const float* xr = x + (size_t)row * EE;
  float s = 0.f;
  for (int e = tid; e < EE; e += 256) s += xr[e];
  red[tid] = s; __syncthreads();
  for (int o = 128; o > 0; o >>= 1) { if (tid < o) red[tid] += red[tid + o]; __syncthreads(); }
  float mu = red[0] * (1.f / EE);
  __syncthreads();
  float v = 0.f;
  for (int e = tid; e < EE; e += 256) { float d = xr[e] - mu; v += d * d; }
  red[tid] = v; __syncthreads();
  for (int o = 128; o > 0; o >>= 1) { if (tid < o) red[tid] += red[tid + o]; __syncthreads(); }
  float rstd = rsqrtf(red[0] * (1.f / EE) + 1e-5f);
  for (int e = tid; e < EE; e += 256)
    out[(size_t)row * EE + e] = (__bf16)((xr[e] - mu) * rstd * w[e] + b[e]);
}

__global__ void zero_loss_k(float* p) { if (threadIdx.x == 0 && blockIdx.x == 0) p[0] = 0.f; }

__global__ __launch_bounds__(256) void loss_rows_k(const float* __restrict__ logits,
                                                   const int* __restrict__ tgt,
                                                   float* __restrict__ rowloss) {
  __shared__ float red[256];
  int row = blockIdx.x, tid = threadIdx.x;
  const float* lg = logits + (size_t)row * VV;
  float m = -1e30f;
  for (int j = tid; j < VV; j += 256) m = fmaxf(m, lg[j]);
  red[tid] = m; __syncthreads();
  for (int o = 128; o > 0; o >>= 1) { if (tid < o) red[tid] = fmaxf(red[tid], red[tid + o]); __syncthreads(); }
  float mx = red[0]; __syncthreads();
  float s = 0.f;
  for (int j = tid; j < VV; j += 256) s += __expf(lg[j] - mx);
  red[tid] = s; __syncthreads();
  for (int o = 128; o > 0; o >>= 1) { if (tid < o) red[tid] += red[tid + o]; __syncthreads(); }
  if (tid == 0) rowloss[row] = mx + logf(red[0]) - lg[tgt[row]];
}

__global__ __launch_bounds__(256) void loss_red_k(const float* __restrict__ rowloss,
                                                  float* __restrict__ lossp) {
  __shared__ float red[256];
  int tid = threadIdx.x;
  float s = 0.f;
  for (int j = tid; j < ROWS; j += 256) s += rowloss[j];
  red[tid] = s; __syncthreads();
  for (int o = 128; o > 0; o >>= 1) { if (tid < o) red[tid] += red[tid + o]; __syncthreads(); }
  if (tid == 0) lossp[0] += red[0] * (1.0f / ROWS);
}

// ---------------------------------------------------------------------------
// Host orchestration
// ---------------------------------------------------------------------------
extern "C" void kernel_launch(void* const* d_in, const int* in_sizes, int n_in,
                              void* d_out, int out_size, void* d_ws, size_t ws_size,
                              hipStream_t stream) {
  const int* tks[LLAYERS + 1];
  for (int i = 0; i <= LLAYERS; i++) tks[i] = (const int*)d_in[i];
  const float* wte    = (const float*)d_in[5];
  const float* wpe    = (const float*)d_in[6];
  const float* ln1_w  = (const float*)d_in[7];
  const float* ln1_b  = (const float*)d_in[8];
  const float* ln2_w  = (const float*)d_in[9];
  const float* ln2_b  = (const float*)d_in[10];
  const float* attn_w = (const float*)d_in[11];
  const float* attn_b = (const float*)d_in[12];
  const float* proj_w = (const float*)d_in[13];
  const float* proj_b = (const float*)d_in[14];
  const float* fc1_w  = (const float*)d_in[15];
  const float* fc1_b  = (const float*)d_in[16];
  const float* fc2_w  = (const float*)d_in[17];
  const float* fc2_b  = (const float*)d_in[18];

  char* ws = (char*)d_ws;
  size_t off = 0;
  auto carve = [&](size_t bytes) -> void* {
    void* p = ws + off;
    off = (off + bytes + 255) & ~(size_t)255;
    return p;
  };
  // weights pre-converted to bf16 and transposed to [N][K]
  __bf16* wte_bf  = (__bf16*)carve((size_t)VV * EE * 2);            // already [N=V][K=E]
  __bf16* awT     = (__bf16*)carve((size_t)LLAYERS * EE * E3 * 2);  // [2304][768]
  __bf16* pwT     = (__bf16*)carve((size_t)LLAYERS * EE * EE * 2);  // [768][768]
  __bf16* f1T     = (__bf16*)carve((size_t)LLAYERS * EE * E3 * 2);  // [2304][768]
  __bf16* f2T     = (__bf16*)carve((size_t)LLAYERS * E3 * EE * 2);  // [768][2304]
  float*  xA      = (float*)carve((size_t)ROWS * EE * 4);
  float*  xB      = (float*)carve((size_t)ROWS * EE * 4);
  __bf16* h_bf    = (__bf16*)carve((size_t)ROWS * EE * 2);
  __bf16* qkv_bf  = (__bf16*)carve((size_t)ROWS * E3 * 2);
  __bf16* y_bf    = (__bf16*)carve((size_t)ROWS * EE * 2);
  __bf16* g_bf    = (__bf16*)carve((size_t)ROWS * E3 * 2);
  __bf16* x3_bf   = (__bf16*)carve((size_t)ROWS * EE * 2);
  float*  rowloss = (float*)carve((size_t)ROWS * 4);

  float* logits = (float*)d_out;
  float* lossp  = (float*)d_out + (size_t)ROWS * VV;

  {  // one-time conversions (per launch)
    long long nwte = (long long)VV * EE;
    cvt_f32_bf16<<<dim3((unsigned)((nwte + 255) / 256)), 256, 0, stream>>>(wte, wte_bf, nwte);
    cvtT_bf16<<<dim3(E3 / 32, EE / 32, LLAYERS), 256, 0, stream>>>(attn_w, awT, EE, E3);
    cvtT_bf16<<<dim3(EE / 32, EE / 32, LLAYERS), 256, 0, stream>>>(proj_w, pwT, EE, EE);
    cvtT_bf16<<<dim3(E3 / 32, EE / 32, LLAYERS), 256, 0, stream>>>(fc1_w, f1T, EE, E3);
    cvtT_bf16<<<dim3(EE / 32, E3 / 32, LLAYERS), 256, 0, stream>>>(fc2_w, f2T, E3, EE);
  }
  zero_loss_k<<<1, 32, 0, stream>>>(lossp);

  for (int i = 0; i < LLAYERS; i++) {
    embed_k<<<ROWS, 256, 0, stream>>>(tks[i], wte, wpe, xA);
    ln_bf16<<<ROWS, 256, 0, stream>>>(xA, ln1_w + (size_t)i * EE, ln1_b + (size_t)i * EE, h_bf);
    // qkv = ln1(x) @ attn_w + attn_b
    gemm_bf16_wmma<<<dim3(E3 / GTN, ROWS / GTM), 256, 0, stream>>>(
        h_bf, EE, awT + (size_t)i * EE * E3, EE,
        attn_b + (size_t)i * E3, nullptr, 0, nullptr, qkv_bf, ROWS, E3, EE, 0);
    attn_wmma<<<dim3(TT / 128, HH, BB), 256, 0, stream>>>(qkv_bf, y_bf);
    // x2 = x + y @ proj_w + proj_b
    gemm_bf16_wmma<<<dim3(EE / GTN, ROWS / GTM), 256, 0, stream>>>(
        y_bf, EE, pwT + (size_t)i * EE * EE, EE,
        proj_b + (size_t)i * EE, xA, EE, xB, nullptr, ROWS, EE, EE, 0);
    ln_bf16<<<ROWS, 256, 0, stream>>>(xB, ln2_w + (size_t)i * EE, ln2_b + (size_t)i * EE, h_bf);
    // g = gelu(ln2(x2) @ fc1_w + fc1_b)
    gemm_bf16_wmma<<<dim3(E3 / GTN, ROWS / GTM), 256, 0, stream>>>(
        h_bf, EE, f1T + (size_t)i * EE * E3, EE,
        fc1_b + (size_t)i * E3, nullptr, 0, nullptr, g_bf, ROWS, E3, EE, 1);
    // x3 = x2 + g @ fc2_w + fc2_b (bf16 only; feeds logits)
    gemm_bf16_wmma<<<dim3(EE / GTN, ROWS / GTM), 256, 0, stream>>>(
        g_bf, E3, f2T + (size_t)i * E3 * EE, E3,
        fc2_b + (size_t)i * EE, xB, EE, nullptr, x3_bf, ROWS, EE, E3, 0);
    // logits = x3 @ wte^T (wte_bf already [V][E] = [N][K])
    gemm_bf16_wmma<<<dim3(VV / GTN, ROWS / GTM), 256, 0, stream>>>(
        x3_bf, EE, wte_bf, EE,
        nullptr, nullptr, 0, logits, nullptr, ROWS, VV, EE, 0);
    loss_rows_k<<<ROWS, 256, 0, stream>>>(logits, tks[i + 1], rowloss);
    loss_red_k<<<1, 256, 0, stream>>>(rowloss, lossp);
  }
}
